// Conv2DMod_32083405701635
// MI455X (gfx1250) — compile-verified, matrix operand order checked by reading, not run on hardware
//
#include <hip/hip_runtime.h>
#include <hip/hip_bf16.h>
#include <stdint.h>

#define B_    16
#define CIN   256
#define COUT  256
#define SDIM  512
#define H_    64
#define W_    64
#define HW    (H_*W_)
#define EPSV  1e-8f
#define LDSK  72   // padded K row stride (halfwords): 36 dwords -> conflict-free b128 frags

typedef __attribute__((ext_vector_type(16))) __bf16 v16bf;
typedef __attribute__((ext_vector_type(8)))  __bf16 v8bf;
typedef __attribute__((ext_vector_type(8)))  float  v8f;

__device__ __forceinline__ unsigned short f2bf(float f) {
  unsigned int u = __float_as_uint(f);
  u += 0x7FFFu + ((u >> 16) & 1u);   // round-to-nearest-even
  return (unsigned short)(u >> 16);
}

// async global->LDS 64B (4 x b128), same base regs, imm offsets (applied to both sides)
__device__ __forceinline__ void async_copy64(unsigned lds, const void* g) {
  asm volatile("global_load_async_to_lds_b128 %0, %1, off"           :: "v"(lds), "v"(g) : "memory");
  asm volatile("global_load_async_to_lds_b128 %0, %1, off offset:16" :: "v"(lds), "v"(g) : "memory");
  asm volatile("global_load_async_to_lds_b128 %0, %1, off offset:32" :: "v"(lds), "v"(g) : "memory");
  asm volatile("global_load_async_to_lds_b128 %0, %1, off offset:48" :: "v"(lds), "v"(g) : "memory");
}

__device__ __forceinline__ void wait_async0() {
  asm volatile("s_wait_asynccnt 0" ::: "memory");
}

// ---------------- style = y @ style_w^T + style_b : [B, CIN] ----------------
__global__ void style_kernel(const float* __restrict__ y,
                             const float* __restrict__ sw,
                             const float* __restrict__ sb,
                             float* __restrict__ style) {
  const int b = blockIdx.x;       // 16 blocks
  const int c = threadIdx.x;      // 256 threads = cin
  const float* yr = y + b * SDIM;
  const float* wr = sw + c * SDIM;
  float acc = 0.f;
  for (int s = 0; s < SDIM; ++s) acc += yr[s] * wr[s];
  style[b * CIN + c] = acc + sb[c];
}

// ---------------- wsq[cout,cin] = sum_tap w^2 ----------------
__global__ void wsq_kernel(const float* __restrict__ w, float* __restrict__ wsq) {
  const int i = blockIdx.x * 256 + threadIdx.x;   // cout*CIN+cin
  const float* p = w + (size_t)i * 9;
  float s = 0.f;
#pragma unroll
  for (int t = 0; t < 9; ++t) { float v = p[t]; s += v * v; }
  wsq[i] = s;
}

// ---------------- wT[tap][cout][cin] = bf16(w[cout][cin][tap]) ----------------
__global__ void wt_kernel(const float* __restrict__ w, unsigned short* __restrict__ wT) {
  const int i = blockIdx.x * 256 + threadIdx.x;   // cout*CIN+cin
#pragma unroll
  for (int t = 0; t < 9; ++t)
    wT[(size_t)t * (COUT * CIN) + i] = f2bf(w[(size_t)i * 9 + t]);
}

// ---------------- d[b,cout] = 1/sqrt(sum_cin wsq*style^2 + eps) ----------------
__global__ void demod_kernel(const float* __restrict__ wsq,
                             const float* __restrict__ style,
                             float* __restrict__ dmod) {
  const int i = blockIdx.x * 256 + threadIdx.x;   // b*COUT+cout
  const int b = i >> 8, co = i & 255;
  const float* wr = wsq + co * CIN;
  const float* sr = style + b * CIN;
  float s = 0.f;
  for (int c = 0; c < CIN; ++c) { float st = sr[c]; s += wr[c] * st * st; }
  dmod[i] = 1.0f / sqrtf(s + EPSV);
}

// ------- xst[b][h][w][cin] = bf16(x[b][cin][h][w] * style[b][cin]) (LDS transpose) -------
__global__ void xprep_kernel(const float* __restrict__ x,
                             const float* __restrict__ style,
                             unsigned short* __restrict__ xst) {
  __shared__ unsigned short tile[64][65];
  const int blk = blockIdx.x;              // B*H*4 blocks
  const int cb = blk & 3;
  const int h  = (blk >> 2) & 63;
  const int b  = blk >> 8;
  const int t  = threadIdx.x;
  const int w  = t & 63;
  const int cg = t >> 6;                   // 0..3
#pragma unroll
  for (int i = 0; i < 16; ++i) {
    const int c = cg * 16 + i;
    const int cin = cb * 64 + c;
    float v = x[(size_t)(b * CIN + cin) * HW + h * W_ + w] * style[b * CIN + cin];
    tile[c][w] = f2bf(v);
  }
  __syncthreads();
  const int wo = t >> 2;
  const int cbase = (t & 3) * 16;
  unsigned int* outp =
      (unsigned int*)(xst + ((size_t)(b * H_ + h) * W_ + wo) * CIN + cb * 64 + cbase);
#pragma unroll
  for (int i = 0; i < 8; ++i) {
    unsigned int lo = tile[cbase + 2 * i][wo];
    unsigned int hi = tile[cbase + 2 * i + 1][wo];
    outp[i] = lo | (hi << 16);
  }
}

// issue the async fills for K-stage s into ping-pong buffer `buf`
__device__ __forceinline__ void fill_stage_async(
    int s, int buf, int b, int m0, int h0, int rfill, int kf,
    const unsigned short* __restrict__ wT,
    const unsigned short* __restrict__ xst,
    unsigned short (*As)[128 * LDSK],
    unsigned short (*Bs)[128 * LDSK]) {
  const int tap = s >> 2;                 // 0..8
  const int cb  = s & 3;                  // cin block
  const int kh = tap / 3, kw = tap % 3;
  const int cin0 = cb << 6;
  // A tile: wT[tap][m0+rfill][cin0+kf .. +31]
  {
    unsigned lds = (unsigned)(uintptr_t)&As[buf][rfill * LDSK + kf];
    const unsigned short* g =
        wT + ((size_t)tap * COUT + m0 + rfill) * CIN + cin0 + kf;
    async_copy64(lds, g);
  }
  // B tile: shifted pixel, zero-fill outside border (async under EXEC / ds_store zeros)
  {
    const int hh = h0 + (rfill >> 6);
    const int ww = rfill & 63;
    const int ph = hh + kh - 1;
    const int pw = ww + kw - 1;
    const bool inb = (ph >= 0) && (ph < H_) && (pw >= 0) && (pw < W_);
    unsigned short* lp = &Bs[buf][rfill * LDSK + kf];
    if (inb) {
      const unsigned short* g =
          xst + ((size_t)(b * H_ + ph) * W_ + pw) * CIN + cin0 + kf;
      async_copy64((unsigned)(uintptr_t)lp, g);
    } else {
      const uint4 z = make_uint4(0u, 0u, 0u, 0u);
      uint4* d = (uint4*)lp;
      d[0] = z; d[1] = z; d[2] = z; d[3] = z;
    }
  }
}

// ---------------- implicit-GEMM modulated conv via bf16 WMMA ----------------
__global__ void __launch_bounds__(256)
conv_kernel(const unsigned short* __restrict__ wT,
            const unsigned short* __restrict__ xst,
            const float* __restrict__ dmod,
            const float* __restrict__ bias,
            float* __restrict__ out) {
  __shared__ unsigned short As[2][128 * LDSK];
  __shared__ unsigned short Bs[2][128 * LDSK];

  const int b  = blockIdx.z;
  const int m0 = blockIdx.y * 128;         // cout tile
  const int n0 = blockIdx.x * 128;         // pixel tile (two rows of W=64)
  const int h0 = n0 >> 6;

  const int t    = threadIdx.x;
  const int wv   = t >> 5;
  const int wm   = wv & 1;                 // wave row (2)
  const int wn   = wv >> 1;                // wave col (4)
  const int lane = t & 31;
  const int half = lane >> 4;
  const int lrow = lane & 15;

  const int rfill = t >> 1;                // 0..127 row to fill
  const int kf    = (t & 1) << 5;          // 0 or 32 (K chunk, halfwords)

  v8f acc[4][2];
#pragma unroll
  for (int i = 0; i < 4; ++i)
#pragma unroll
    for (int j = 0; j < 2; ++j)
      acc[i][j] = (v8f){0.f, 0.f, 0.f, 0.f, 0.f, 0.f, 0.f, 0.f};

  // prologue: stage 0 in flight
  fill_stage_async(0, 0, b, m0, h0, rfill, kf, wT, xst, As, Bs);

  for (int s = 0; s < 36; ++s) {
    wait_async0();          // own stage-s transfers resident in LDS
    __syncthreads();        // all waves' stage-s resident; all done reading buf[(s+1)&1]
    if (s < 35)             // overlap next stage's DMA with this stage's math
      fill_stage_async(s + 1, (s + 1) & 1, b, m0, h0, rfill, kf, wT, xst, As, Bs);

    const unsigned short* Ab = As[s & 1];
    const unsigned short* Bb = Bs[s & 1];
#pragma unroll
    for (int ks = 0; ks < 2; ++ks) {
      v16bf af[4];
      v16bf bfv[2];
#pragma unroll
      for (int fm = 0; fm < 4; ++fm) {
        const int off = (wm * 64 + fm * 16 + lrow) * LDSK + ks * 32 + half * 8;
        v8bf lo = *(const v8bf*)&Ab[off];
        v8bf hi = *(const v8bf*)&Ab[off + 16];
        af[fm] = __builtin_shufflevector(lo, hi, 0, 1, 2, 3, 4, 5, 6, 7,
                                         8, 9, 10, 11, 12, 13, 14, 15);
      }
#pragma unroll
      for (int fn = 0; fn < 2; ++fn) {
        const int off = (wn * 32 + fn * 16 + lrow) * LDSK + ks * 32 + half * 8;
        v8bf lo = *(const v8bf*)&Bb[off];
        v8bf hi = *(const v8bf*)&Bb[off + 16];
        bfv[fn] = __builtin_shufflevector(lo, hi, 0, 1, 2, 3, 4, 5, 6, 7,
                                          8, 9, 10, 11, 12, 13, 14, 15);
      }
#pragma unroll
      for (int fm = 0; fm < 4; ++fm)
#pragma unroll
        for (int fn = 0; fn < 2; ++fn)
          acc[fm][fn] = __builtin_amdgcn_wmma_f32_16x16x32_bf16(
              false, af[fm], false, bfv[fn], (short)0, acc[fm][fn], false, false);
    }
  }

  // Epilogue: demodulate + bias; C layout: VGPR r -> M = r + half*8, N = lane&15
#pragma unroll
  for (int fm = 0; fm < 4; ++fm) {
#pragma unroll
    for (int fn = 0; fn < 2; ++fn) {
      const int np = n0 + wn * 32 + fn * 16 + lrow;
#pragma unroll
      for (int r = 0; r < 8; ++r) {
        const int co = m0 + wm * 64 + fm * 16 + half * 8 + r;
        const float dsc = dmod[b * COUT + co];
        const float bs  = bias[co];
        out[(size_t)(b * COUT + co) * HW + np] = acc[fm][fn][r] * dsc + bs;
      }
    }
  }
}

extern "C" void kernel_launch(void* const* d_in, const int* in_sizes, int n_in,
                              void* d_out, int out_size, void* d_ws, size_t ws_size,
                              hipStream_t stream) {
  const float* x    = (const float*)d_in[0];
  const float* y    = (const float*)d_in[1];
  const float* w    = (const float*)d_in[2];
  const float* bias = (const float*)d_in[3];
  const float* sw   = (const float*)d_in[4];
  const float* sb   = (const float*)d_in[5];
  float* out = (float*)d_out;

  // workspace layout (all regions fully rewritten every call)
  float* style = (float*)d_ws;                         // 16*256
  float* wsq   = style + B_ * CIN;                     // 256*256
  float* dmod  = wsq + COUT * CIN;                     // 16*256
  unsigned short* wT  = (unsigned short*)(dmod + B_ * COUT);   // 9*256*256 bf16
  unsigned short* xst = wT + (size_t)9 * COUT * CIN;           // 16*64*64*256 bf16

  style_kernel<<<B_, CIN, 0, stream>>>(y, sw, sb, style);
  wsq_kernel<<<(COUT * CIN) / 256, 256, 0, stream>>>(w, wsq);
  wt_kernel<<<(COUT * CIN) / 256, 256, 0, stream>>>(w, wT);
  demod_kernel<<<(B_ * COUT) / 256, 256, 0, stream>>>(wsq, style, dmod);
  xprep_kernel<<<B_ * H_ * 4, 256, 0, stream>>>(x, style, xst);

  dim3 grid(HW / 128, COUT / 128, B_);
  conv_kernel<<<grid, 256, 0, stream>>>(wT, xst, dmod, bias, out);
}
